// MultiHeadGraphAttentionLayer_33552284516832
// MI455X (gfx1250) — matrix operand order repaired and checked by reading, not compiled, not run on hardware
//
#include <hip/hip_runtime.h>
#include <hip/hip_bf16.h>

typedef __attribute__((ext_vector_type(16))) _Float16 v16h;
typedef __attribute__((ext_vector_type(2)))  _Float16 h2;
typedef __attribute__((ext_vector_type(8)))  float    v8f;
typedef __attribute__((ext_vector_type(4)))  int      v4i;

#define NNODES 8192
#define INFEAT 256
#define OUTF   64
#define NHEADS 4
#define GALPHA 0.2f
#define LOG2E  1.44269504f
#define JSTAGE 256                      // adjacency columns staged per LDS buffer
#define NSTAGES (NNODES / JSTAGE)       // 32
#define LPAD   (JSTAGE + 4)             // pad rows to 260 ints: uniform 2-way banks

#if defined(__AMDGCN__) && \
    __has_builtin(__builtin_amdgcn_global_load_async_to_lds_b128) && \
    __has_builtin(__builtin_amdgcn_s_wait_asynccnt)
#define USE_ASYNC 1
#else
#define USE_ASYNC 0
#endif

#if __has_builtin(__builtin_elementwise_maximumnum)
#define H2MAX(a, b) __builtin_elementwise_maximumnum((a), (b))
#else
#define H2MAX(a, b) __builtin_elementwise_max((a), (b))
#endif

// ---------------------------------------------------------------------------
// Kernel 1: per-head projection Wh = h @ W[h]; store transposed f16 WhT[h][f][i]
// plus attention logits pre-scaled by log2(e):  src = (Wh.a1)*log2e (f32),
// dst = (Wh.a2)*log2e as f32 (for max-reduce) and f16 (hot-loop operand).
// ---------------------------------------------------------------------------
__global__ __launch_bounds__(64) void k_proj(const float* __restrict__ hin,
                                             const float* __restrict__ W,
                                             const float* __restrict__ a1,
                                             const float* __restrict__ a2,
                                             _Float16* __restrict__ whT,
                                             float* __restrict__ srcv,
                                             float* __restrict__ dstv,
                                             _Float16* __restrict__ dstH) {
  const int i  = blockIdx.x;
  const int hd = blockIdx.y;
  const int o  = threadIdx.x;  // 0..63
  const float* hr = hin + (size_t)i * INFEAT;
  const float* wp = W + (size_t)hd * INFEAT * OUTF + o;
  float acc = 0.f;
#pragma unroll 4
  for (int k = 0; k < INFEAT; ++k) acc += hr[k] * wp[(size_t)k * OUTF];
  whT[(size_t)(hd * OUTF + o) * NNODES + i] = (_Float16)acc;

  float s1 = acc * a1[hd * OUTF + o];
  float s2 = acc * a2[hd * OUTF + o];
  for (int off = 16; off > 0; off >>= 1) {
    s1 += __shfl_down(s1, off);
    s2 += __shfl_down(s2, off);
  }
  __shared__ float r1[2], r2[2];
  if ((o & 31) == 0) { r1[o >> 5] = s1; r2[o >> 5] = s2; }
  __syncthreads();
  if (o == 0) {
    const float sv = (r1[0] + r1[1]) * LOG2E;
    const float dv = (r2[0] + r2[1]) * LOG2E;
    srcv[hd * NNODES + i] = sv;
    dstv[hd * NNODES + i] = dv;
    dstH[hd * NNODES + i] = (_Float16)dv;
  }
}

// ---------------------------------------------------------------------------
// Kernel 2: maxd[h] = max_j dst[h][j]
// ---------------------------------------------------------------------------
__global__ __launch_bounds__(256) void k_maxdst(const float* __restrict__ dstv,
                                                float* __restrict__ maxd) {
  const int hd = blockIdx.x;
  float m = -1e30f;
  for (int j = threadIdx.x; j < NNODES; j += 256)
    m = fmaxf(m, dstv[hd * NNODES + j]);
  for (int off = 16; off > 0; off >>= 1) m = fmaxf(m, __shfl_xor(m, off));
  __shared__ float red[8];
  if ((threadIdx.x & 31) == 0) red[threadIdx.x >> 5] = m;
  __syncthreads();
  if (threadIdx.x == 0) {
    float mm = red[0];
    for (int w = 1; w < 8; ++w) mm = fmaxf(mm, red[w]);
    maxd[hd] = mm;
  }
}

// ---------------------------------------------------------------------------
// Kernel 3: fused masked-softmax attention + AV GEMM. Block = 16 query rows,
// 4 waves = 4 heads. Adjacency tile (shared by all heads) staged through a
// double-buffered LDS tile with GLOBAL_LOAD_ASYNC_TO_LDS, synced via
// s_wait_asynccnt + workgroup split-barrier. Row sums via a 5th WMMA against
// an all-ones B. Masking exploits adj in {0,1}: f16 mask pair is
// ((a1<<16)|a0)*0x3C00 — no compares, applied after exp2 (arg<=0 so exp2<=1).
// ---------------------------------------------------------------------------
__global__ __launch_bounds__(128) void k_attn(const int* __restrict__ adj,
                                              const _Float16* __restrict__ whT,
                                              const float* __restrict__ srcv,
                                              const _Float16* __restrict__ dstH,
                                              const float* __restrict__ maxd,
                                              float* __restrict__ concat) {
  const int hd   = threadIdx.x >> 5;          // wave id = head
  const int lane = threadIdx.x & 31;
  const int ln   = lane & 15;                 // row (A frag) or column (B/D frag)
  const int hs   = lane >> 4;                 // lane-half select
  const int hs8  = hs * 8;
  const int i0   = blockIdx.x * 16;

  const float srcR = srcv[hd * NNODES + i0 + ln];   // already *log2e
  float c = srcR + maxd[hd];
  c = c > 0.f ? c : GALPHA * c;               // lrelu of unmasked row max

  // lrelu(s+d)-c == max(d + (s-c), 0.2d + (0.2s-c)); precompute the constants
  const _Float16 scH  = (_Float16)(srcR - c);
  const _Float16 ascH = (_Float16)(GALPHA * srcR - c);
  const h2 scv    = {scH, scH};
  const h2 ascv   = {ascH, ascH};
  const h2 alpha2 = {(_Float16)GALPHA, (_Float16)GALPHA};

  const _Float16* dh  = dstH + (size_t)hd * NNODES;
  const _Float16* bwh = whT  + (size_t)hd * OUTF * NNODES;

  v16h onesv;
#pragma unroll
  for (int e = 0; e < 16; ++e) onesv[e] = (_Float16)1.0f;

  v8f accs[4];
#pragma unroll
  for (int fb = 0; fb < 4; ++fb) accs[fb] = (v8f){0.f,0.f,0.f,0.f,0.f,0.f,0.f,0.f};
  v8f accL = (v8f){0.f,0.f,0.f,0.f,0.f,0.f,0.f,0.f};   // row-sum tile

  // one 32-column chunk: batched loads, masked-softmax A fragment, 5 WMMAs
  auto process32 = [&](const int* ajs, int j0) {
    v16h Bf[4];                       // load all B fragments first (clause-able)
#pragma unroll
    for (int fb = 0; fb < 4; ++fb) {
      union { int4 q[2]; v16h v; } B;
      const int4* bp =
          (const int4*)(bwh + (size_t)(fb * 16 + ln) * NNODES + j0 + hs8);
      B.q[0] = bp[0]; B.q[1] = bp[2];
      Bf[fb] = B.v;
    }
    union { int4 q[2]; h2 d[8]; } D2;
    D2.q[0] = *(const int4*)(dh + j0 + hs8);
    D2.q[1] = *(const int4*)(dh + j0 + 16 + hs8);

    union { v16h v; h2 p2[8]; } A;
#pragma unroll
    for (int e = 0; e < 8; ++e) {
      h2 u   = D2.d[e] + scv;                    // d + (s-c)
      h2 w   = D2.d[e] * alpha2 + ascv;          // 0.2d + (0.2s-c)  (pk_fma)
      h2 arg = H2MAX(u, w);                      // = lrelu(s+d)-c  <= 0
      h2 p   = __builtin_elementwise_exp2(arg);  // v_exp_f16 x2, <= 1
      unsigned m = (((unsigned)ajs[2 * e + 1]) << 16) | (unsigned)ajs[2 * e];
      m *= 0x3C00u;                              // 0/1 -> f16 0.0/1.0 pair
      A.p2[e] = p * __builtin_bit_cast(h2, m);   // v_pk_mul_f16 mask
    }

    accL = __builtin_amdgcn_wmma_f32_16x16x32_f16(
        false, A.v, false, onesv, (short)0, accL, false, false);
#pragma unroll
    for (int fb = 0; fb < 4; ++fb)
      accs[fb] = __builtin_amdgcn_wmma_f32_16x16x32_f16(
          false, A.v, false, Bf[fb], (short)0, accs[fb], false, false);
  };

#if USE_ASYNC
  __shared__ int ladj[2][16][LPAD];
  const int tid   = threadIdx.x;
  const int arow  = tid >> 3;        // 0..15: adjacency row this thread stages
  const int acol0 = (tid & 7) * 4;   // 16B-aligned column group

  auto issue_stage = [&](int s, int buf) {
    const int* g = adj + (size_t)(i0 + arow) * NNODES + s * JSTAGE + acol0;
    int* l = &ladj[buf][arow][acol0];
#pragma unroll
    for (int q = 0; q < 8; ++q)
      __builtin_amdgcn_global_load_async_to_lds_b128(
          (__attribute__((address_space(1))) v4i*)(g + q * 32),
          (__attribute__((address_space(3))) v4i*)(l + q * 32), 0, 0);
  };

  issue_stage(0, 0);
  for (int s = 0; s < NSTAGES; ++s) {
    const int buf = s & 1;
    __builtin_amdgcn_s_wait_asynccnt(0);   // my stage-s tiles landed
    __syncthreads();                       // everyone's landed; prev buf consumed
    if (s + 1 < NSTAGES) issue_stage(s + 1, buf ^ 1);
    for (int ck = 0; ck < JSTAGE / 32; ++ck) {
      union { int4 q[4]; int w[16]; } aj;
      const int4* lpa = (const int4*)&ladj[buf][ln][ck * 32 + hs8];
      const int4* lpb = (const int4*)&ladj[buf][ln][ck * 32 + 16 + hs8];
      aj.q[0] = lpa[0]; aj.q[1] = lpa[1];
      aj.q[2] = lpb[0]; aj.q[3] = lpb[1];
      process32(aj.w, s * JSTAGE + ck * 32);
    }
  }
#else
  const int* ap = adj + (size_t)(i0 + ln) * NNODES;
  for (int j0 = 0; j0 < NNODES; j0 += 32) {
    union { int4 q[4]; int w[16]; } aj;
    const int4* ap4a = (const int4*)(ap + j0 + hs8);
    const int4* ap4b = (const int4*)(ap + j0 + 16 + hs8);
    aj.q[0] = ap4a[0]; aj.q[1] = ap4a[1];
    aj.q[2] = ap4b[0]; aj.q[3] = ap4b[1];
    if (j0 + 32 < NNODES) __builtin_prefetch(ap + j0 + 32, 0, 1);
    process32(aj.w, j0);
  }
#endif

  // normalize by row sums (already in matching rows of accL), ELU, scatter
#pragma unroll
  for (int r = 0; r < 8; ++r) {
#if defined(__AMDGCN__)
    const float linv = __builtin_amdgcn_rcpf(accL[r]);
#else
    const float linv = 1.0f / accL[r];
#endif
#pragma unroll
    for (int fb = 0; fb < 4; ++fb) {
      float v = accs[fb][r] * linv;
      v = v > 0.f ? v : (__expf(v) - 1.0f);
      concat[(size_t)(i0 + r + hs8) * (NHEADS * OUTF) + hd * OUTF + fb * 16 + ln] = v;
    }
  }
}

// ---------------------------------------------------------------------------
// Kernel 4: out = concat @ fc_w^T + fc_b via WMMA f16 (K=256).
// ---------------------------------------------------------------------------
__global__ __launch_bounds__(128) void k_fc(const float* __restrict__ concat,
                                            const float* __restrict__ fcw,
                                            const float* __restrict__ fcb,
                                            float* __restrict__ out) {
  const int wv   = threadIdx.x >> 5;
  const int lane = threadIdx.x & 31;
  const int ln   = lane & 15;
  const int hs   = lane >> 4;
  const int hs8  = hs * 8;
  const int i0   = blockIdx.x * 16;
  const int ob   = wv * 16;

  const float* arow = concat + (size_t)(i0 + ln) * (NHEADS * OUTF);
  const float* brow = fcw + (size_t)(ob + ln) * (NHEADS * OUTF);

  v8f acc = (v8f){0.f,0.f,0.f,0.f,0.f,0.f,0.f,0.f};
  for (int k0 = 0; k0 < NHEADS * OUTF; k0 += 32) {
    union { float4 q[4]; float s[16]; } av, bv;
    const float4* a4a = (const float4*)(arow + k0 + hs8);
    const float4* a4b = (const float4*)(arow + k0 + 16 + hs8);
    av.q[0] = a4a[0]; av.q[1] = a4a[1]; av.q[2] = a4b[0]; av.q[3] = a4b[1];
    const float4* b4 = (const float4*)(brow + k0 + hs8);
    bv.q[0] = b4[0]; bv.q[1] = b4[1]; bv.q[2] = b4[4]; bv.q[3] = b4[5];

    union { v16h v; _Float16 e[16]; } A, B;
#pragma unroll
    for (int e = 0; e < 16; ++e) {
      A.e[e] = (_Float16)av.s[e];
      B.e[e] = (_Float16)bv.s[e];
    }
    acc = __builtin_amdgcn_wmma_f32_16x16x32_f16(
        false, A.v, false, B.v, (short)0, acc, false, false);
  }
  const float bias = fcb[ob + ln];
#pragma unroll
  for (int r = 0; r < 8; ++r)
    out[(size_t)(i0 + r + hs8) * OUTF + ob + ln] = acc[r] + bias;
}

// ---------------------------------------------------------------------------
extern "C" void kernel_launch(void* const* d_in, const int* in_sizes, int n_in,
                              void* d_out, int out_size, void* d_ws, size_t ws_size,
                              hipStream_t stream) {
  const float* hin = (const float*)d_in[0];
  const int*   adj = (const int*)d_in[1];
  const float* W   = (const float*)d_in[2];
  const float* a1  = (const float*)d_in[3];
  const float* a2  = (const float*)d_in[4];
  const float* fcw = (const float*)d_in[5];
  const float* fcb = (const float*)d_in[6];
  float* out = (float*)d_out;

  char* ws = (char*)d_ws;
  // workspace layout (~12.4 MB total)
  _Float16* whT  = (_Float16*)(ws);                                  // 4 MB
  float*    srcv = (float*)(ws + (4u << 20));                        // 128 KB
  float*    dstv = (float*)(ws + (4u << 20) + (128u << 10));         // 128 KB
  float*    maxd = (float*)(ws + (4u << 20) + (256u << 10));         // 16 B
  _Float16* dstH = (_Float16*)(ws + (4u << 20) + (257u << 10));      // 64 KB
  float*    conc = (float*)(ws + (4u << 20) + (321u << 10));         // 8 MB

  k_proj  <<<dim3(NNODES, NHEADS), 64, 0, stream>>>(hin, W, a1, a2, whT,
                                                    srcv, dstv, dstH);
  k_maxdst<<<NHEADS, 256, 0, stream>>>(dstv, maxd);
  k_attn  <<<NNODES / 16, 128, 0, stream>>>(adj, whT, srcv, dstH, maxd, conc);
  k_fc    <<<NNODES / 16, 128, 0, stream>>>(conc, fcw, fcb, out);
}
